// BahdanauAttentionMechanism_66443144069732
// MI455X (gfx1250) — matrix operand order repaired and checked by reading, not compiled, not run on hardware
//
#include <hip/hip_runtime.h>
#include <hip/hip_bf16.h>
#include <math.h>

// ---------------------------------------------------------------------------
// Problem constants
// ---------------------------------------------------------------------------
#define BB 32
#define TT 4096
#define AD 256
#define VD 256
#define QD 256

// TDM-staged tile: 64 rows x 256 f32, padded 4 DWORDs/row -> 260-dword pitch
#define TILE_ROWS 64
#define ROW_PITCH 260   // 256 + 4 pad DWORDs (bank-conflict-free f32 reads)
#define NWAVE 4         // 128 threads / block

typedef __attribute__((ext_vector_type(16))) __bf16 v16bf;
typedef __attribute__((ext_vector_type(8)))  float  v8f;
typedef __attribute__((ext_vector_type(4)))  unsigned int u32x4;
typedef __attribute__((ext_vector_type(8)))  int          i32x8;

__device__ __forceinline__ float fast_tanh(float x) {
#if __has_builtin(__builtin_amdgcn_tanhf)
    return __builtin_amdgcn_tanhf(x);   // v_tanh_f32 (TRANS, co-executes w/ WMMA)
#else
    return tanhf(x);
#endif
}

__device__ __forceinline__ void wait_tensorcnt0() {
#if __has_builtin(__builtin_amdgcn_s_wait_tensorcnt)
    __builtin_amdgcn_s_wait_tensorcnt(0);
#else
    asm volatile("s_wait_tensorcnt 0x0" ::: "memory");
#endif
}

// ---------------------------------------------------------------------------
// Kernel 1: qb[b,a] = sum_q Wq[a,q]*query[b,q] + bq[a] + bv[a]
// ---------------------------------------------------------------------------
__global__ void prep_qb_kernel(const float* __restrict__ query,
                               const float* __restrict__ Wq,
                               const float* __restrict__ bq,
                               const float* __restrict__ bv,
                               float* __restrict__ qb) {
    int b = blockIdx.x;
    int a = threadIdx.x;
    const float* q = query + (size_t)b * QD;
    const float* w = Wq + (size_t)a * QD;
    float s = 0.f;
#pragma unroll 8
    for (int i = 0; i < QD; ++i) s += w[i] * q[i];
    qb[(size_t)b * AD + a] = s + bq[a] + bv[a];
}

// ---------------------------------------------------------------------------
// Kernel 2: Wv f32 -> bf16 row-major (== WMMA B-fragment layout)
// ---------------------------------------------------------------------------
__global__ void conv_wv_kernel(const float* __restrict__ Wv,
                               __bf16* __restrict__ wvb) {
    int i = blockIdx.x * 256 + threadIdx.x;
    wvb[i] = (__bf16)Wv[i];
}

// ---------------------------------------------------------------------------
// Kernel 3 (hot): fused keys-GEMM + tanh + score dot.
// grid = B*(T/64), 128 threads (4 waves), wave owns a 16-row strip.
// values tile staged into LDS by the Tensor Data Mover (tensor_load_to_lds);
// each wave then converts its rows to bf16 ONCE into a fragment-major LDS
// region (lane's 16 bf16 contiguous), so an A fragment is a single aligned
// 32-byte LDS read with zero VALU — cheap whether the compiler keeps
// fragments resident or rematerializes them.  B fragments double-buffered
// one N-tile ahead of the 8 back-to-back v_wmma_f32_16x16x32_bf16.
// ---------------------------------------------------------------------------
__global__ void __launch_bounds__(128)
score_kernel(const float* __restrict__ values,
             const __bf16* __restrict__ wvb,
             const float* __restrict__ qb,
             const float* __restrict__ wv_score,
             const float* __restrict__ bv_score,
             float* __restrict__ e_ws) {
    const int blk   = blockIdx.x;
    const int b     = blk >> 6;          // 64 chunks per batch
    const int chunk = blk & 63;
    const int t0    = chunk * TILE_ROWS;
    const int tid   = threadIdx.x;
    const int w     = tid >> 5;          // wave 0..3
    const int lane  = tid & 31;
    const int hl    = lane >> 4;
    const int l16   = lane & 15;

    __shared__ float tile[TILE_ROWS * ROW_PITCH];      // 65 KB raw f32 tile
    __shared__ v16bf atile[NWAVE * 8 * 32];            // 32 KB bf16 fragments

    // ---- Issue TDM load of the 64x256 f32 tile (wave 0 only) --------------
    if (w == 0) {
        const unsigned lds_off =
            (unsigned)(unsigned long long)(uintptr_t)&tile[0];  // LDS byte offset
        const unsigned long long ga =
            (unsigned long long)(uintptr_t)(values + ((size_t)b * TT + t0) * VD);

        // D# group 0: count=1 | lds_addr | global_addr | type=2
        u32x4 g0;
        g0[0] = 1u;                                   // count=1, no gather
        g0[1] = lds_off;                              // lds_addr (bytes)
        g0[2] = (unsigned)(ga & 0xFFFFFFFFull);       // global_addr[31:0]
        g0[3] = (unsigned)((ga >> 32) & 0x01FFFFFFull) | 0x80000000u; // type=2

        // D# group 1: data_size=4B, pad_enable, pad_interval=256dw, pad=4dw
        i32x8 g1;
        g1[0] = (int)((2u << 16) | (1u << 20) | (7u << 22) | (3u << 25));
        g1[1] = (int)((256u & 0xFFFFu) << 16);        // tensor_dim0 = 256 (lo)
        g1[2] = (int)((unsigned)TILE_ROWS << 16);     // dim0 hi=0 | tensor_dim1 lo
        g1[3] = (int)(256u << 16);                    // dim1 hi=0 | tile_dim0=256
        g1[4] = TILE_ROWS;                            // tile_dim1=64, tile_dim2=0
        g1[5] = 256;                                  // tensor_dim0_stride = 256
        g1[6] = 0;
        g1[7] = 0;

        asm volatile("tensor_load_to_lds %0, %1" :: "s"(g0), "s"(g1) : "memory");
        wait_tensorcnt0();
    }

    // ---- per-lane q/w scalars: one per N-tile, preloaded into registers ---
    float qreg[16], wreg[16];
#pragma unroll
    for (int nt = 0; nt < 16; ++nt) {
        qreg[nt] = qb[(size_t)b * AD + nt * 16 + l16];
        wreg[nt] = wv_score[nt * 16 + l16];
    }

    __syncthreads();   // TDM tile visible to all waves

    // ---- one-time f32 -> bf16 fragment conversion into LDS ----------------
    // A 16x32 layout (contiguous k, matching B): lanes 0-15 (row=l16) hold K {0..15},
    // lanes 16-31 hold K {16..31}.  Row pitch 260 dwords (260%64==4)
    // keeps the 16 rows of a half-wave on distinct 4-bank groups.
    {
        const float* arow = &tile[(w * 16 + l16) * ROW_PITCH];
#pragma unroll
        for (int kt = 0; kt < 8; ++kt) {
            const float* p = arow + kt * 32 + (hl ? 16 : 0);
            v16bf f;
#pragma unroll
            for (int j = 0; j < 8; ++j) {
                f[j]     = (__bf16)p[j];
                f[8 + j] = (__bf16)p[8 + j];
            }
            atile[(w * 8 + kt) * 32 + lane] = f;   // fragment-major, 32B/lane
        }
    }

    float e_part[8];
#pragma unroll
    for (int r = 0; r < 8; ++r) e_part[r] = 0.f;

    // ---- A fragments: single 32B LDS read each (resident or cheap remat) --
    const v16bf* afr = &atile[w * 8 * 32 + lane];
    v16bf afrag[8];
#pragma unroll
    for (int kt = 0; kt < 8; ++kt)
        afrag[kt] = afr[kt * 32];

    // ---- 16 N-tiles of AD; B fragments double-buffered one tile ahead ----
    const __bf16* bbase = wvb + (size_t)l16 * VD + (hl ? 16 : 0);
    v16bf bfA[8], bfB[8];
#pragma unroll
    for (int kt = 0; kt < 8; ++kt)
        bfA[kt] = *(const v16bf*)(bbase + kt * 32);

#pragma unroll
    for (int nt = 0; nt < 16; ++nt) {
        v16bf* cur = (nt & 1) ? bfB : bfA;
        v16bf* nxt = (nt & 1) ? bfA : bfB;
        if (nt < 15) {
            const __bf16* bc = bbase + (size_t)(nt + 1) * 16 * VD;
#pragma unroll
            for (int kt = 0; kt < 8; ++kt)
                nxt[kt] = *(const v16bf*)(bc + kt * 32);
        }
        v8f acc = {};
#pragma unroll
        for (int kt = 0; kt < 8; ++kt)
            acc = __builtin_amdgcn_wmma_f32_16x16x32_bf16(
                false, afrag[kt], false, cur[kt], (short)0, acc, false, false);

        // C layout: VGPR r -> row r / r+8 (half), lane%16 -> column.
        // Per-lane (per-column) partials; cross-lane reduce once at the end.
#pragma unroll
        for (int r = 0; r < 8; ++r)
            e_part[r] += fast_tanh(qreg[nt] + acc[r]) * wreg[nt];
    }

    // ---- single 16-lane butterfly reduction per row -----------------------
#pragma unroll
    for (int r = 0; r < 8; ++r) {
        float tv = e_part[r];
        tv += __shfl_xor(tv, 1, 32);
        tv += __shfl_xor(tv, 2, 32);
        tv += __shfl_xor(tv, 4, 32);
        tv += __shfl_xor(tv, 8, 32);
        e_part[r] = tv;
    }
    if (l16 == 0) {
        const float bvs = bv_score[0];
        const int rbase = t0 + w * 16 + hl * 8;
#pragma unroll
        for (int r = 0; r < 8; ++r)
            e_ws[(size_t)b * TT + rbase + r] = e_part[r] + bvs;
    }
}

// ---------------------------------------------------------------------------
// Kernel 4: per-batch softmax over T=4096 -> a output
// ---------------------------------------------------------------------------
__global__ void softmax_kernel(const float* __restrict__ e_ws,
                               float* __restrict__ a_out) {
    const int b   = blockIdx.x;
    const int tid = threadIdx.x;
    __shared__ float red[256];
    const float* e = e_ws + (size_t)b * TT;

    float v[16];
    float m = -INFINITY;
#pragma unroll
    for (int i = 0; i < 16; ++i) { v[i] = e[tid + i * 256]; m = fmaxf(m, v[i]); }
    red[tid] = m; __syncthreads();
    for (int s = 128; s > 0; s >>= 1) {
        if (tid < s) red[tid] = fmaxf(red[tid], red[tid + s]);
        __syncthreads();
    }
    m = red[0]; __syncthreads();

    float l = 0.f;
#pragma unroll
    for (int i = 0; i < 16; ++i) { v[i] = __expf(v[i] - m); l += v[i]; }
    red[tid] = l; __syncthreads();
    for (int s = 128; s > 0; s >>= 1) {
        if (tid < s) red[tid] += red[tid + s];
        __syncthreads();
    }
    const float inv = 1.f / red[0];
#pragma unroll
    for (int i = 0; i < 16; ++i)
        a_out[(size_t)b * TT + tid + i * 256] = v[i] * inv;
}

// ---------------------------------------------------------------------------
// Kernel 5a: zero the context accumulator (d_out is poisoned 0xAA)
// ---------------------------------------------------------------------------
__global__ void zero_c_kernel(float* __restrict__ c_out) {
    c_out[blockIdx.x * 256 + threadIdx.x] = 0.f;
}

// ---------------------------------------------------------------------------
// Kernel 5b: c[b,v] = sum_t a[b,t]*values[b,t,v], T split 32-way per batch,
// merged with global f32 atomics (global_atomic_add_f32).
// ---------------------------------------------------------------------------
__global__ void __launch_bounds__(256)
context_kernel(const float* __restrict__ values,
               const float* __restrict__ a,
               float* __restrict__ c_out) {
    const int b     = blockIdx.x >> 5;
    const int chunk = blockIdx.x & 31;
    const int tid   = threadIdx.x;
    const int t0    = chunk * 128;
    const float* vb = values + ((size_t)b * TT + t0) * VD + tid;
    const float* ab = a + (size_t)b * TT + t0;
    float acc = 0.f;
#pragma unroll 4
    for (int tt = 0; tt < 128; ++tt) {
        __builtin_prefetch(vb + (size_t)(tt + 8) * VD, 0, 3);
        acc = fmaf(ab[tt], vb[(size_t)tt * VD], acc);
    }
    atomicAdd(&c_out[b * VD + tid], acc);
}

// ---------------------------------------------------------------------------
// Host launcher  (inputs in setup_inputs order; out = (c, a) concatenated)
// ---------------------------------------------------------------------------
extern "C" void kernel_launch(void* const* d_in, const int* in_sizes, int n_in,
                              void* d_out, int out_size, void* d_ws, size_t ws_size,
                              hipStream_t stream) {
    const float* query    = (const float*)d_in[0];
    const float* values   = (const float*)d_in[1];
    const float* Wq       = (const float*)d_in[2];
    const float* bq       = (const float*)d_in[3];
    const float* Wv       = (const float*)d_in[4];
    const float* bv       = (const float*)d_in[5];
    const float* wv_score = (const float*)d_in[6];
    const float* bv_score = (const float*)d_in[7];

    float* c_out = (float*)d_out;                       // B*VD
    float* a_out = (float*)d_out + (size_t)BB * VD;     // B*T

    char* ws = (char*)d_ws;
    float*  qb   = (float*) (ws + 0);                   // 32 KB
    __bf16* wvb  = (__bf16*)(ws + 32 * 1024);           // 128 KB
    float*  e_ws = (float*) (ws + 160 * 1024);          // 512 KB

    prep_qb_kernel<<<BB, AD, 0, stream>>>(query, Wq, bq, bv, qb);
    conv_wv_kernel<<<(AD * VD) / 256, 256, 0, stream>>>(Wv, wvb);
    score_kernel<<<BB * (TT / TILE_ROWS), 128, 0, stream>>>(values, wvb, qb,
                                                            wv_score, bv_score, e_ws);
    softmax_kernel<<<BB, 256, 0, stream>>>(e_ws, a_out);
    zero_c_kernel<<<BB, 256, 0, stream>>>(c_out);
    context_kernel<<<BB * 32, 256, 0, stream>>>(values, a_out, c_out);
}
